// TransformerXLEncoderLayer_31439160606837
// MI455X (gfx1250) — matrix operand order repaired and checked by reading, not compiled, not run on hardware
//
#include <hip/hip_runtime.h>
#include <cstdint>

typedef __bf16 bf16;
typedef __attribute__((ext_vector_type(16))) __bf16 v16bf;
typedef __attribute__((ext_vector_type(8)))  float  v8f;
typedef __attribute__((ext_vector_type(4)))  unsigned int v4u;

union FragBF { v16bf v; v4u u[2]; };

__device__ __forceinline__ unsigned short f2bfu(float f) {
  union { float f; unsigned u; } a; a.f = f;
  unsigned r = (a.u + 0x7FFFu + ((a.u >> 16) & 1u)) >> 16;
  return (unsigned short)r;
}
__device__ __forceinline__ bf16 f2bf(float f) {
  union { unsigned short s; bf16 b; } o; o.s = f2bfu(f); return o.b;
}

// Low 32 bits of a flat pointer into LDS == LDS byte address
__device__ __forceinline__ unsigned lds_addr_of(const void* p) {
  return (unsigned)(uintptr_t)p;
}

// CDNA5 async global->LDS copies (ASYNCcnt), GV addressing mode.
// INST_OFFSET is added to BOTH the LDS and the global address (ISA 08 §4.4),
// so the +16B variant reuses the same base registers.
__device__ __forceinline__ void async_ld_b128(unsigned lds, const bf16* g) {
  asm volatile("global_load_async_to_lds_b128 %0, %1, off"
               :: "v"(lds), "v"((unsigned long long)(uintptr_t)g)
               : "memory");
}
__device__ __forceinline__ void async_ld_b128_off16(unsigned lds, const bf16* g) {
  asm volatile("global_load_async_to_lds_b128 %0, %1, off offset:16"
               :: "v"(lds), "v"((unsigned long long)(uintptr_t)g)
               : "memory");
}
__device__ __forceinline__ void wait_async_le3() {
  asm volatile("s_wait_asynccnt 3" ::: "memory");
}
__device__ __forceinline__ void wait_async_0() {
  asm volatile("s_wait_asynccnt 0" ::: "memory");
}

constexpr int CB = 2, CS = 2048, CD = 1024, CH = 16, CHD = 64, CFF = 4096;
constexpr int CPREV = 1024, CT = 3072;   // total keys = PREV + S == cache_limit (no trim)

// ---------------- f32 -> bf16 convert ----------------
__global__ void k_f32_to_bf16(const float* __restrict__ src,
                              unsigned short* __restrict__ dst, int n) {
  int i = blockIdx.x * blockDim.x + threadIdx.x;
  if (i < n) dst[i] = f2bfu(src[i]);
}

// ---------------- RMSNorm (one block per row of 1024) ----------------
__global__ __launch_bounds__(256)
void k_rmsnorm(const float* __restrict__ x, const float* __restrict__ w,
               unsigned short* __restrict__ out) {
  int row = blockIdx.x;
  const float4* xr = (const float4*)(x + (size_t)row * CD);
  float4 v = xr[threadIdx.x];
  float ss = v.x * v.x + v.y * v.y + v.z * v.z + v.w * v.w;
#pragma unroll
  for (int m = 16; m >= 1; m >>= 1) ss += __shfl_xor(ss, m, 32);
  __shared__ float red[8];
  if ((threadIdx.x & 31) == 0) red[threadIdx.x >> 5] = ss;
  __syncthreads();
  float tot = 0.f;
#pragma unroll
  for (int i = 0; i < 8; i++) tot += red[i];
  float sc = rsqrtf(tot / (float)CD + 1e-6f);
  float4 ww = ((const float4*)w)[threadIdx.x];
  unsigned short* o = out + (size_t)row * CD + threadIdx.x * 4;
  o[0] = f2bfu(v.x * sc * ww.x);
  o[1] = f2bfu(v.y * sc * ww.y);
  o[2] = f2bfu(v.z * sc * ww.z);
  o[3] = f2bfu(v.w * sc * ww.w);
}

// ---------------- WMMA GEMM: C[M,N] = A[M,K] @ B[N,K]^T + bias ----------------
// Async-pipelined: double-buffered LDS tiles filled by global_load_async_to_lds_b128,
// last chunk peeled so the steady-state loop is branch-free.
// mode 0: f32 out = acc+bias
// mode 2: f32 out = acc+bias + aux[idx]           (residual)
// mode 3: bf16 out = silu(aux[idx]) * (acc+bias)  (SwiGLU fusion)
__global__ __launch_bounds__(256)
void k_gemm(const bf16* __restrict__ A, const bf16* __restrict__ Bm,
            const float* __restrict__ bias, const float* __restrict__ aux,
            void* __restrict__ C, int M, int N, int K, int mode) {
  __shared__ bf16 sA[2][128 * 40];
  __shared__ bf16 sB[2][64 * 40];
  const int tid = threadIdx.x;
  const int bm = blockIdx.y * 128, bn = blockIdx.x * 64;
  const int wave = tid >> 5, lane = tid & 31, lr = lane & 15, lp = lane >> 4;
  const int wm = (wave >> 1) * 32, wn = (wave & 1) * 32;
  const int arow = tid >> 1, acol = (tid & 1) * 16;   // A tile: 128 x 32
  const int brow = tid >> 2, bcol = (tid & 3) * 8;    // B tile:  64 x 32

  // Per-thread source rows / LDS destinations (3 async b128 per chunk).
  const bf16* gA = A + (size_t)(bm + arow) * K + acol;
  const bf16* gB = Bm + (size_t)(bn + brow) * K + bcol;
  unsigned ldsA[2], ldsB[2];
  ldsA[0] = lds_addr_of(&sA[0][arow * 40 + acol]);
  ldsA[1] = lds_addr_of(&sA[1][arow * 40 + acol]);
  ldsB[0] = lds_addr_of(&sB[0][brow * 40 + bcol]);
  ldsB[1] = lds_addr_of(&sB[1][brow * 40 + bcol]);

  v8f zf = {};
  v8f acc[2][2];
  acc[0][0] = zf; acc[0][1] = zf; acc[1][0] = zf; acc[1][1] = zf;

  auto issue = [&](int chunk, int buf) {
    const bf16* na = gA + (size_t)chunk * 32;
    const bf16* nb = gB + (size_t)chunk * 32;
    async_ld_b128(ldsA[buf], na);
    async_ld_b128_off16(ldsA[buf], na);   // same bases, offset:16 on LDS+global
    async_ld_b128(ldsB[buf], nb);
  };
  auto compute = [&](int buf) {
    FragBF fa[2], fb[2];
#pragma unroll
    for (int ii = 0; ii < 2; ii++) {      // A layout: contiguous 16-K run [16lp..]
      const bf16* p = sA[buf] + (wm + 16 * ii + lr) * 40;
      fa[ii].u[0] = *(const v4u*)(p + 16 * lp);
      fa[ii].u[1] = *(const v4u*)(p + 16 * lp + 8);
    }
#pragma unroll
    for (int j = 0; j < 2; j++) {         // B layout: lane=N col, contiguous 16-K run
      const bf16* p = sB[buf] + (wn + 16 * j + lr) * 40 + 16 * lp;
      fb[j].u[0] = *(const v4u*)p;
      fb[j].u[1] = *(const v4u*)(p + 8);
    }
#pragma unroll
    for (int ii = 0; ii < 2; ii++)
#pragma unroll
      for (int j = 0; j < 2; j++)
        acc[ii][j] = __builtin_amdgcn_wmma_f32_16x16x32_bf16(
            false, fa[ii].v, false, fb[j].v, (short)0, acc[ii][j], false, false);
  };

  const int nchunks = K >> 5;             // >= 2 for all our shapes
  issue(0, 0);                            // prologue: chunk 0 -> buf 0
  int i = 0;
  for (; i < nchunks - 1; ++i) {          // branch-free steady state
    issue(i + 1, (i + 1) & 1);
    wait_async_le3();                     // current buffer's 3 loads have landed
    __syncthreads();
    compute(i & 1);
    __syncthreads();                      // buf free before it is re-issued at i+2
  }
  wait_async_0();                         // epilogue: last chunk
  __syncthreads();
  compute(i & 1);

#pragma unroll
  for (int ii = 0; ii < 2; ii++) {
#pragma unroll
    for (int j = 0; j < 2; j++) {
      int col = bn + wn + 16 * j + lr;
      float bs = bias ? bias[col] : 0.0f;
#pragma unroll
      for (int r = 0; r < 8; r++) {
        int row = bm + wm + 16 * ii + 8 * lp + r;
        size_t idx = (size_t)row * N + col;
        float v = acc[ii][j][r] + bs;
        if (mode == 0) {
          ((float*)C)[idx] = v;
        } else if (mode == 2) {
          ((float*)C)[idx] = v + aux[idx];
        } else {
          float g = aux[idx];
          float sg = g / (1.0f + __expf(-g));
          ((unsigned short*)C)[idx] = f2bfu(sg * v);
        }
      }
    }
  }
}

// ---------------- RoPE + pack new q/k/v into head-major bf16 ----------------
__global__ void k_pack_new(const float* __restrict__ qf, const float* __restrict__ kf,
                           const float* __restrict__ vf, const float* __restrict__ cosb,
                           const float* __restrict__ sinb,
                           unsigned short* __restrict__ q_ro,
                           unsigned short* __restrict__ k_ro,
                           unsigned short* __restrict__ vT) {
  int idx = blockIdx.x * blockDim.x + threadIdx.x;   // [0, B*S*H*32)
  int u = idx & 31;
  int h = (idx >> 5) & 15;
  int s = (idx >> 9) & 2047;
  int b = idx >> 20;
  int pos = CPREV + s;                // query pos == new-key pos
  float c = cosb[pos * 32 + u], sn = sinb[pos * 32 + u];
  size_t src = ((size_t)(b * CS + s)) * CD + h * CHD + 2 * u;
  float q0 = qf[src], q1 = qf[src + 1];
  float k0 = kf[src], k1 = kf[src + 1];
  float v0 = vf[src], v1 = vf[src + 1];
  size_t bh = (size_t)(b * CH + h);
  size_t qo = (bh * CS + s) * CHD + 2 * u;
  q_ro[qo]     = f2bfu(q0 * c - q1 * sn);
  q_ro[qo + 1] = f2bfu(q0 * sn + q1 * c);
  int t = CPREV + s;
  size_t ko = (bh * CT + t) * CHD + 2 * u;
  k_ro[ko]     = f2bfu(k0 * c - k1 * sn);
  k_ro[ko + 1] = f2bfu(k0 * sn + k1 * c);
  size_t vo = (bh * CHD + 2 * u) * CT + t;   // V transposed: [B,H,64,T]
  vT[vo]      = f2bfu(v0);
  vT[vo + CT] = f2bfu(v1);
}

// ---------------- RoPE + pack cached k/v ----------------
__global__ void k_pack_cache(const float* __restrict__ ck, const float* __restrict__ cv,
                             const float* __restrict__ cosb, const float* __restrict__ sinb,
                             unsigned short* __restrict__ k_ro,
                             unsigned short* __restrict__ vT) {
  int idx = blockIdx.x * blockDim.x + threadIdx.x;   // [0, B*H*PREV*32)
  int u = idx & 31;
  int t = (idx >> 5) & 1023;
  int h = (idx >> 15) & 15;
  int b = idx >> 19;
  size_t bh = (size_t)(b * CH + h);
  size_t src = (bh * CPREV + t) * CHD + 2 * u;
  float k0 = ck[src], k1 = ck[src + 1];
  float v0 = cv[src], v1 = cv[src + 1];
  float c = cosb[t * 32 + u], sn = sinb[t * 32 + u];
  size_t ko = (bh * CT + t) * CHD + 2 * u;
  k_ro[ko]     = f2bfu(k0 * c - k1 * sn);
  k_ro[ko + 1] = f2bfu(k0 * sn + k1 * c);
  size_t vo = (bh * CHD + 2 * u) * CT + t;
  vT[vo]      = f2bfu(v0);
  vT[vo + CT] = f2bfu(v1);
}

// ---------------- Flash attention: wave per 16-query tile ----------------
__global__ __launch_bounds__(256)
void k_attn(const bf16* __restrict__ q, const bf16* __restrict__ k,
            const bf16* __restrict__ vT, unsigned short* __restrict__ ctx) {
  __shared__ bf16 sP[8][16 * 40];
  const int wave = threadIdx.x >> 5, lane = threadIdx.x & 31;
  const int lr = lane & 15, lp = lane >> 4;
  const int gid = blockIdx.x * 8 + wave;
  const int qt = gid & 127;            // S/16 = 128 tiles
  const int bh = gid >> 7;
  const int h = bh & 15, b = bh >> 4;
  const int qs = qt * 16;
  const float scale = 0.125f;          // 1/sqrt(64)
  const float slope = 2.0f / (float)(CT - 1);

  const bf16* qbase = q + ((size_t)(b * CH + h) * CS + qs) * CHD;
  FragBF fq[2];
#pragma unroll
  for (int kc = 0; kc < 2; kc++) {
    const bf16* p = qbase + lr * CHD + kc * 32;
    fq[kc].u[0] = *(const v4u*)(p + 16 * lp);
    fq[kc].u[1] = *(const v4u*)(p + 16 * lp + 8);
  }

  v8f zf = {};
  v8f o[4]; o[0] = zf; o[1] = zf; o[2] = zf; o[3] = zf;
  float mrow[8], lrow[8];
#pragma unroll
  for (int r = 0; r < 8; r++) { mrow[r] = -1e30f; lrow[r] = 0.f; }

  const bf16* kbh = k + (size_t)(b * CH + h) * CT * CHD;
  const bf16* vbh = vT + (size_t)(b * CH + h) * CHD * CT;
  const int kend = CPREV + qs + 16;    // exclusive causal key bound for this tile

  for (int t0 = 0; t0 < kend; t0 += 32) {
    v8f s0 = zf, s1 = zf;
#pragma unroll
    for (int kc = 0; kc < 2; kc++) {
      FragBF fk0, fk1;
      const bf16* p0 = kbh + (size_t)(t0 + lr) * CHD + kc * 32 + 16 * lp;
      fk0.u[0] = *(const v4u*)p0;
      fk0.u[1] = *(const v4u*)(p0 + 8);
      const bf16* p1 = p0 + 16 * CHD;
      fk1.u[0] = *(const v4u*)p1;
      fk1.u[1] = *(const v4u*)(p1 + 8);
      s0 = __builtin_amdgcn_wmma_f32_16x16x32_bf16(false, fq[kc].v, false, fk0.v,
                                                   (short)0, s0, false, false);
      s1 = __builtin_amdgcn_wmma_f32_16x16x32_bf16(false, fq[kc].v, false, fk1.v,
                                                   (short)0, s1, false, false);
    }
#pragma unroll
    for (int r = 0; r < 8; r++) {
      int m = 8 * lp + r;
      int iabs = CPREV + qs + m;
      float sc0 = s0[r] * scale + slope * (float)(t0 + lr - iabs);
      float sc1 = s1[r] * scale + slope * (float)(t0 + 16 + lr - iabs);
      if (t0 + lr > iabs)      sc0 = -1e30f;   // causal
      if (t0 + 16 + lr > iabs) sc1 = -1e30f;
      float vmax = fmaxf(sc0, sc1);
#pragma unroll
      for (int msk = 8; msk >= 1; msk >>= 1)
        vmax = fmaxf(vmax, __shfl_xor(vmax, msk, 32));
      float mnew = fmaxf(mrow[r], vmax);
      float alpha = __expf(mrow[r] - mnew);
      float p0 = __expf(sc0 - mnew);
      float p1 = __expf(sc1 - mnew);
      float rs = p0 + p1;
#pragma unroll
      for (int msk = 8; msk >= 1; msk >>= 1) rs += __shfl_xor(rs, msk, 32);
      lrow[r] = lrow[r] * alpha + rs;
      mrow[r] = mnew;
      o[0][r] *= alpha; o[1][r] *= alpha; o[2][r] *= alpha; o[3][r] *= alpha;
      sP[wave][m * 40 + lr] = f2bf(p0);
      sP[wave][m * 40 + 16 + lr] = f2bf(p1);
    }
    FragBF fp;   // reload P in A-operand layout (per-wave LDS, in-order DS ops)
    {
      const bf16* pp = &sP[wave][lr * 40];
      fp.u[0] = *(const v4u*)(pp + 16 * lp);
      fp.u[1] = *(const v4u*)(pp + 16 * lp + 8);
    }
#pragma unroll
    for (int jf = 0; jf < 4; jf++) {
      FragBF fv;
      const bf16* pv = vbh + (size_t)(16 * jf + lr) * CT + t0 + 16 * lp;
      fv.u[0] = *(const v4u*)pv;
      fv.u[1] = *(const v4u*)(pv + 8);
      o[jf] = __builtin_amdgcn_wmma_f32_16x16x32_bf16(false, fp.v, false, fv.v,
                                                      (short)0, o[jf], false, false);
    }
  }

#pragma unroll
  for (int r = 0; r < 8; r++) {
    float inv = 1.0f / lrow[r];
    int m = 8 * lp + r;
    size_t rowbase = ((size_t)(b * CS + qs + m)) * CD + h * CHD;
    ctx[rowbase + 0  + lr] = f2bfu(o[0][r] * inv);
    ctx[rowbase + 16 + lr] = f2bfu(o[1][r] * inv);
    ctx[rowbase + 32 + lr] = f2bfu(o[2][r] * inv);
    ctx[rowbase + 48 + lr] = f2bfu(o[3][r] * inv);
  }
}

extern "C" void kernel_launch(void* const* d_in, const int* in_sizes, int n_in,
                              void* d_out, int out_size, void* d_ws, size_t ws_size,
                              hipStream_t stream) {
  (void)in_sizes; (void)n_in; (void)out_size; (void)ws_size;
  const float* x       = (const float*)d_in[0];
  const float* cache_k = (const float*)d_in[1];
  const float* cache_v = (const float*)d_in[2];
  const float* cosb    = (const float*)d_in[3];
  const float* sinb    = (const float*)d_in[4];
  const float* attn_w  = (const float*)d_in[5];
  const float* ffn_w   = (const float*)d_in[6];
  const float* wq = (const float*)d_in[7];  const float* bq = (const float*)d_in[8];
  const float* wk = (const float*)d_in[9];  const float* bk = (const float*)d_in[10];
  const float* wv = (const float*)d_in[11]; const float* bv = (const float*)d_in[12];
  const float* wo = (const float*)d_in[13]; const float* bo = (const float*)d_in[14];
  const float* wg = (const float*)d_in[15]; const float* bg = (const float*)d_in[16];
  const float* wvl= (const float*)d_in[17]; const float* bvl= (const float*)d_in[18];
  const float* wp = (const float*)d_in[19]; const float* bp = (const float*)d_in[20];

  char* base = (char*)d_ws;
  size_t off = 0;
  auto carve = [&](size_t bytes) -> char* {
    char* p = base + off;
    off += (bytes + 255) & ~(size_t)255;
    return p;
  };
  const int NTOK = CB * CS;  // 4096 rows

  unsigned short* wq_b  = (unsigned short*)carve((size_t)CD * CD * 2);
  unsigned short* wk_b  = (unsigned short*)carve((size_t)CD * CD * 2);
  unsigned short* wv_b  = (unsigned short*)carve((size_t)CD * CD * 2);
  unsigned short* wo_b  = (unsigned short*)carve((size_t)CD * CD * 2);
  unsigned short* wg_b  = (unsigned short*)carve((size_t)CFF * CD * 2);
  unsigned short* wvl_b = (unsigned short*)carve((size_t)CFF * CD * 2);
  unsigned short* wp_b  = (unsigned short*)carve((size_t)CD * CFF * 2);
  unsigned short* hbuf  = (unsigned short*)carve((size_t)NTOK * CD * 2);
  float* qf   = (float*)carve((size_t)NTOK * CD * 4);
  float* kf   = (float*)carve((size_t)NTOK * CD * 4);
  float* vf   = (float*)carve((size_t)NTOK * CD * 4);
  unsigned short* q_ro = (unsigned short*)carve((size_t)CB * CH * CS * CHD * 2);
  unsigned short* k_ro = (unsigned short*)carve((size_t)CB * CH * CT * CHD * 2);
  unsigned short* vTb  = (unsigned short*)carve((size_t)CB * CH * CT * CHD * 2);
  unsigned short* ctxb = (unsigned short*)carve((size_t)NTOK * CD * 2);
  float* xat  = (float*)carve((size_t)NTOK * CD * 4);
  float* gbuf = (float*)carve((size_t)NTOK * CFF * 4);
  unsigned short* gvb = (unsigned short*)carve((size_t)NTOK * CFF * 2);

  auto conv = [&](const float* s, unsigned short* d, int n) {
    k_f32_to_bf16<<<(n + 255) / 256, 256, 0, stream>>>(s, d, n);
  };
  conv(wq, wq_b, CD * CD);   conv(wk, wk_b, CD * CD);
  conv(wv, wv_b, CD * CD);   conv(wo, wo_b, CD * CD);
  conv(wg, wg_b, CFF * CD);  conv(wvl, wvl_b, CFF * CD);
  conv(wp, wp_b, CD * CFF);

  // h = rmsnorm(x) -> bf16
  k_rmsnorm<<<NTOK, 256, 0, stream>>>(x, attn_w, hbuf);

  // q,k,v projections (f32 out for RoPE precision)
  dim3 gqkv(CD / 64, NTOK / 128);
  k_gemm<<<gqkv, 256, 0, stream>>>((const bf16*)hbuf, (const bf16*)wq_b, bq, nullptr, qf, NTOK, CD, CD, 0);
  k_gemm<<<gqkv, 256, 0, stream>>>((const bf16*)hbuf, (const bf16*)wk_b, bk, nullptr, kf, NTOK, CD, CD, 0);
  k_gemm<<<gqkv, 256, 0, stream>>>((const bf16*)hbuf, (const bf16*)wv_b, bv, nullptr, vf, NTOK, CD, CD, 0);

  // RoPE + pack (head-major bf16; V transposed)
  k_pack_new<<<(CB * CS * CH * 32) / 256, 256, 0, stream>>>(qf, kf, vf, cosb, sinb, q_ro, k_ro, vTb);
  k_pack_cache<<<(CB * CH * CPREV * 32) / 256, 256, 0, stream>>>(cache_k, cache_v, cosb, sinb, k_ro, vTb);

  // attention -> ctx (bf16)
  k_attn<<<(CB * CH * (CS / 16)) / 8, 256, 0, stream>>>((const bf16*)q_ro, (const bf16*)k_ro,
                                                        (const bf16*)vTb, ctxb);

  // x_attn = x + ctx @ wo^T + bo
  k_gemm<<<gqkv, 256, 0, stream>>>((const bf16*)ctxb, (const bf16*)wo_b, bo, x, xat, NTOK, CD, CD, 2);

  // h2 = rmsnorm(x_attn)
  k_rmsnorm<<<NTOK, 256, 0, stream>>>(xat, ffn_w, hbuf);

  // gate (f32), then value-GEMM with fused silu(gate)*value -> bf16
  dim3 gff(CFF / 64, NTOK / 128);
  k_gemm<<<gff, 256, 0, stream>>>((const bf16*)hbuf, (const bf16*)wg_b, bg, nullptr, gbuf, NTOK, CFF, CD, 0);
  k_gemm<<<gff, 256, 0, stream>>>((const bf16*)hbuf, (const bf16*)wvl_b, bvl, gbuf, gvb, NTOK, CFF, CD, 3);

  // out = x_attn + gv @ w_proj^T + b_proj
  dim3 gpr(CD / 64, NTOK / 128);
  k_gemm<<<gpr, 256, 0, stream>>>((const bf16*)gvb, (const bf16*)wp_b, bp, xat, d_out, NTOK, CD, CFF, 2);
}
